// MoTBlock_50319836840363
// MI455X (gfx1250) — matrix operand order repaired and checked by reading, not compiled, not run on hardware
//
#include <hip/hip_runtime.h>
#include <hip/hip_bf16.h>
#include <cstdint>

typedef __attribute__((ext_vector_type(16))) _Float16 v16h;
typedef __attribute__((ext_vector_type(8)))  _Float16 v8h;
typedef __attribute__((ext_vector_type(8)))  float    v8f;
typedef __attribute__((ext_vector_type(4)))  unsigned int v4u;
typedef __attribute__((ext_vector_type(8)))  int      v8i;
typedef __attribute__((ext_vector_type(4)))  int      v4i;

#define BN_INV 0.999995f   // 1/sqrt(1+1e-5)

__device__ __forceinline__ v8f wmma32(v16h a, v16h b, v8f c) {
    // D = A(16x32 f16) * B(32x16 f16) + C(16x16 f32)
    return __builtin_amdgcn_wmma_f32_16x16x32_f16(false, a, false, b, (short)0, c, false, false);
}

// K-index inside a 16x32 f16 A fragment for element i (0..15) of lane-group g (0/1)
__device__ __forceinline__ int aK(int i, int g) {
    return (g << 4) | i;
}

// ---- CDNA5 async global->LDS copy (ASYNCcnt), 16 bytes per lane ----
__device__ __forceinline__ void async_copy16(const void* g, void* l) {
    uint32_t laddr = (uint32_t)(uintptr_t)l;     // low 32 bits of generic LDS ptr = LDS offset
    uint64_t gaddr = (uint64_t)(uintptr_t)g;
    asm volatile("global_load_async_to_lds_b128 %0, %1, off"
                 :: "v"(laddr), "v"(gaddr) : "memory");
}
__device__ __forceinline__ void async_wait0() {
    asm volatile("s_wait_asynccnt 0" ::: "memory");
}

// ---- CDNA5 Tensor Data Mover: 1-D contiguous global->LDS DMA (TENSORcnt) ----
// Descriptor per CDNA5 ISA: data_size=8B, count=1, type=2.
__device__ __forceinline__ void tdm_load_1d(const void* gsrc, void* ldst, unsigned bytes) {
    uint32_t lds = (uint32_t)(uintptr_t)ldst;
    uint64_t ga  = (uint64_t)(uintptr_t)gsrc;
    uint32_t n8  = bytes >> 3;                   // 8-byte units (<= 65535)
    v4u g0;
    g0[0] = 1u;                                                  // count=1, user mode
    g0[1] = lds;                                                 // lds_addr
    g0[2] = (uint32_t)ga;                                        // global_addr[31:0]
    g0[3] = (uint32_t)((ga >> 32) & 0x1FFFFFFu) | (2u << 30);    // global_addr[56:32] | type=2
    v8i g1;
    g1[0] = 0x30000;                                             // data_size=3 (8B), no flags
    g1[1] = (int)((n8 & 0xFFFFu) << 16);                         // tensor_dim0 lo16 (bits 63:48)
    g1[2] = (int)(((n8 >> 16) & 0xFFFFu) | (1u << 16));          // tensor_dim0 hi16 | tensor_dim1=1
    g1[3] = (int)((n8 & 0xFFFFu) << 16);                         // tensor_dim1 hi (0) | tile_dim0
    g1[4] = 1;                                                   // tile_dim1=1, tile_dim2=0
    g1[5] = (int)n8;                                             // tensor_dim0_stride lo32
    g1[6] = 0;
    g1[7] = 0;
    v4i z4 = {0, 0, 0, 0};
#if __has_include(<hip/amd_detail/amd_gfx1250_TDM.h>)
    v8i z8 = {0, 0, 0, 0, 0, 0, 0, 0};
    __builtin_amdgcn_tensor_load_to_lds(g0, g1, z4, z4, z8, 0);  // clang-23 6-arg form
#else
    __builtin_amdgcn_tensor_load_to_lds(g0, g1, z4, z4, 0);      // ROCm 7.2 5-arg form
#endif
}

// ---------------------------------------------------------------------------
// Kernel 1: per-(b,c) 64x64 attention + depthwise conv + BN, fp16 output
// ---------------------------------------------------------------------------
__global__ __launch_bounds__(256) void k_attn(const float* __restrict__ x,
                                              const float* __restrict__ dw_w,
                                              const float* __restrict__ dw_g,
                                              const float* __restrict__ dw_b,
                                              _Float16* __restrict__ a_out) {
    __shared__ float    Xf[64 * 65];
    __shared__ _Float16 Xh[64 * 72];
    __shared__ float    Gs[64 * 65];
    __shared__ _Float16 Ph[64 * 72];
    __shared__ float    kw[9];

    const int bc = blockIdx.x;          // b*128 + c
    const int c  = bc & 127;
    const float* xp = x + (size_t)bc * 4096;
    const int t = threadIdx.x;

    // load 64x64 plane (f32 + f16 copies, padded rows); conversion needs VGPR path
    #pragma unroll
    for (int it = 0; it < 4; ++it) {
        int idx = t * 4 + it * 1024;
        float4 v = *reinterpret_cast<const float4*>(xp + idx);
        int r = idx >> 6, cc = idx & 63;
        Xf[r * 65 + cc + 0] = v.x; Xf[r * 65 + cc + 1] = v.y;
        Xf[r * 65 + cc + 2] = v.z; Xf[r * 65 + cc + 3] = v.w;
        Xh[r * 72 + cc + 0] = (_Float16)v.x; Xh[r * 72 + cc + 1] = (_Float16)v.y;
        Xh[r * 72 + cc + 2] = (_Float16)v.z; Xh[r * 72 + cc + 3] = (_Float16)v.w;
    }
    if (t < 9) kw[t] = dw_w[c * 9 + t];
    __syncthreads();

    const int wv = t >> 5, lane = t & 31;
    const int g = lane >> 4, n = lane & 15;

    // ---- G = X * X^T  (scale 1/8) ----
    #pragma unroll
    for (int ph = 0; ph < 2; ++ph) {
        int tile = wv + ph * 8;
        int mi = tile >> 2, ni = tile & 3;
        v8f acc;
        #pragma unroll
        for (int r = 0; r < 8; ++r) acc[r] = 0.0f;
        #pragma unroll
        for (int kk = 0; kk < 64; kk += 32) {
            v16h af, bf;
            const _Float16* arow = Xh + (mi * 16 + (lane & 15)) * 72 + kk;
            const _Float16* brow = Xh + (ni * 16 + n) * 72 + kk + g * 16;  // B[k][n]=X[n][k]
            #pragma unroll
            for (int i = 0; i < 16; ++i) { af[i] = arow[aK(i, g)]; bf[i] = brow[i]; }
            acc = wmma32(af, bf, acc);
        }
        #pragma unroll
        for (int r = 0; r < 8; ++r) {
            int m = mi * 16 + r + (g << 3);
            Gs[m * 65 + ni * 16 + n] = acc[r] * 0.125f;
        }
    }
    __syncthreads();

    // ---- row softmax -> P (f16) ----
    if (t < 64) {
        float mx = -1e30f;
        for (int v = 0; v < 64; ++v) mx = fmaxf(mx, Gs[t * 65 + v]);
        float s = 0.0f;
        for (int v = 0; v < 64; ++v) s += __expf(Gs[t * 65 + v] - mx);
        float inv = 1.0f / s;
        for (int v = 0; v < 64; ++v)
            Ph[t * 72 + v] = (_Float16)(__expf(Gs[t * 65 + v] - mx) * inv);
    }
    __syncthreads();

    const float gsc = dw_g[c] * BN_INV;
    const float gbi = dw_b[c];

    // ---- A = P * X  + bn(dwconv(x)) ----
    #pragma unroll
    for (int ph = 0; ph < 2; ++ph) {
        int tile = wv + ph * 8;
        int mi = tile >> 2, ni = tile & 3;
        v8f acc;
        #pragma unroll
        for (int r = 0; r < 8; ++r) acc[r] = 0.0f;
        #pragma unroll
        for (int kk = 0; kk < 64; kk += 32) {
            v16h af, bf;
            const _Float16* arow = Ph + (mi * 16 + (lane & 15)) * 72 + kk;
            #pragma unroll
            for (int i = 0; i < 16; ++i) {
                af[i] = arow[aK(i, g)];
                bf[i] = Xh[(kk + g * 16 + i) * 72 + ni * 16 + n];   // B[k][n]=X[k][n]
            }
            acc = wmma32(af, bf, acc);
        }
        #pragma unroll
        for (int r = 0; r < 8; ++r) {
            int h = mi * 16 + r + (g << 3);
            int w = ni * 16 + n;
            float dv = 0.0f;
            #pragma unroll
            for (int dy = -1; dy <= 1; ++dy) {
                int hh = h + dy;
                if (hh < 0 || hh > 63) continue;
                #pragma unroll
                for (int dx = -1; dx <= 1; ++dx) {
                    int ww = w + dx;
                    if (ww < 0 || ww > 63) continue;
                    dv += Xf[hh * 65 + ww] * kw[(dy + 1) * 3 + dx + 1];
                }
            }
            float val = acc[r] + dv * gsc + gbi;
            a_out[(size_t)bc * 4096 + h * 64 + w] = (_Float16)val;
        }
    }
}

// ---------------------------------------------------------------------------
// Kernel 2: ghost primary 1x1 conv GEMM  out = relu(bn(W @ in)), optional SE gate
// Weights staged via TDM (TENSORcnt); activations via async-to-LDS (ASYNCcnt).
// ---------------------------------------------------------------------------
__global__ __launch_bounds__(256) void k_gemm(const _Float16* __restrict__ in,
                                              const _Float16* __restrict__ wh,
                                              const float* __restrict__ gamma,
                                              const float* __restrict__ beta,
                                              const float* __restrict__ gate,
                                              float* __restrict__ out,
                                              int IC, int OC) {
    extern __shared__ char smem[];
    _Float16* Wsh = (_Float16*)smem;        // OC*IC
    _Float16* Ash = Wsh + OC * IC;          // IC*128

    const int b = blockIdx.x >> 5;
    const int p0 = (blockIdx.x & 31) * 128;
    const int t = threadIdx.x;

    // stage weights with the Tensor Data Mover (one DMA, issued by wave 0)
    if (t < 32) {
        tdm_load_1d(wh, Wsh, (unsigned)(OC * IC) * 2u);
    }

    // stage activation chunk
    const int na = IC * 16;
    if (gate) {
        // SE gate folded in -> VGPR convert path
        for (int v = t; v < na; v += 256) {
            int ic = v >> 4, off = (v & 15) * 8;
            v8h d = *reinterpret_cast<const v8h*>(in + ((size_t)b * IC + ic) * 4096 + p0 + off);
            float gt = gate[b * IC + ic];
            #pragma unroll
            for (int j = 0; j < 8; ++j) d[j] = (_Float16)((float)d[j] * gt);
            *reinterpret_cast<v8h*>(Ash + ic * 128 + off) = d;
        }
    } else {
        // raw copy -> async global->LDS, no VGPR round-trip
        for (int v = t; v < na; v += 256) {
            int ic = v >> 4, off = (v & 15) * 8;
            async_copy16(in + ((size_t)b * IC + ic) * 4096 + p0 + off,
                         Ash + ic * 128 + off);
        }
        async_wait0();
    }
    if (t < 32) __builtin_amdgcn_s_wait_tensorcnt(0);
    __syncthreads();

    const int wv = t >> 5, lane = t & 31;
    const int g = lane >> 4, n = lane & 15;
    const int pl = wv * 16;                 // this wave's pixel tile
    const int mts = OC >> 4;

    for (int mt = 0; mt < mts; ++mt) {
        v8f acc;
        #pragma unroll
        for (int r = 0; r < 8; ++r) acc[r] = 0.0f;
        for (int kk = 0; kk < IC; kk += 32) {
            v16h af, bf;
            const _Float16* arow = Wsh + (mt * 16 + (lane & 15)) * IC + kk;
            #pragma unroll
            for (int i = 0; i < 16; ++i) {
                af[i] = arow[aK(i, g)];
                bf[i] = Ash[(kk + g * 16 + i) * 128 + pl + n];
            }
            acc = wmma32(af, bf, acc);
        }
        #pragma unroll
        for (int r = 0; r < 8; ++r) {
            int oc = mt * 16 + r + (g << 3);
            float val = acc[r] * gamma[oc] * BN_INV + beta[oc];
            val = fmaxf(val, 0.0f);
            out[((size_t)b * OC + oc) * 4096 + p0 + pl + n] = val;
        }
    }
}

// ---------------------------------------------------------------------------
// Kernel 3: ghost cheap dw3x3 + BN + ReLU + concat + residual (+f16, +SE sums)
// Plane staged via async global->LDS (pad 68 keeps 16B alignment per row chunk).
// ---------------------------------------------------------------------------
__global__ __launch_bounds__(256) void k_cheap(const float* __restrict__ x1,
                                               const float* __restrict__ cw,
                                               const float* __restrict__ cg,
                                               const float* __restrict__ cb,
                                               int halfc,
                                               const float* __restrict__ res,
                                               float res_scale,
                                               float* __restrict__ outf,
                                               _Float16* __restrict__ outh,
                                               float* __restrict__ sums) {
    __shared__ float P[64 * 68];
    __shared__ float kw[9];
    __shared__ float red[256];

    const int bc = blockIdx.x;
    const int b = bc / halfc, ch = bc % halfc;
    const float* src = x1 + (size_t)bc * 4096;
    const int t = threadIdx.x;

    #pragma unroll
    for (int it = 0; it < 4; ++it) {
        int idx = t * 4 + it * 1024;
        int r = idx >> 6, cc = idx & 63;
        async_copy16(src + idx, &P[r * 68 + cc]);
    }
    if (t < 9) kw[t] = cw[ch * 9 + t];
    async_wait0();
    __syncthreads();

    const float sc = cg[ch] * BN_INV, sb = cb[ch];
    const int CH2 = halfc * 2;
    const size_t base1 = ((size_t)b * CH2 + ch) * 4096;
    const size_t base2 = ((size_t)b * CH2 + halfc + ch) * 4096;

    float s1 = 0.0f, s2 = 0.0f;
    #pragma unroll
    for (int i = 0; i < 16; ++i) {
        int px = t + i * 256;
        int h = px >> 6, w = px & 63;
        float x1v = P[h * 68 + w];
        float d = 0.0f;
        #pragma unroll
        for (int dy = -1; dy <= 1; ++dy) {
            int hh = h + dy;
            if (hh < 0 || hh > 63) continue;
            #pragma unroll
            for (int dx = -1; dx <= 1; ++dx) {
                int ww = w + dx;
                if (ww < 0 || ww > 63) continue;
                d += P[hh * 68 + ww] * kw[(dy + 1) * 3 + dx + 1];
            }
        }
        float x2v = fmaxf(d * sc + sb, 0.0f);
        float o1 = x1v + (res ? res_scale * res[base1 + px] : 0.0f);
        float o2 = x2v + (res ? res_scale * res[base2 + px] : 0.0f);
        if (outf) { outf[base1 + px] = o1; outf[base2 + px] = o2; }
        if (outh) { outh[base1 + px] = (_Float16)o1; outh[base2 + px] = (_Float16)o2; }
        s1 += o1; s2 += o2;
    }

    if (sums) {
        red[t] = s1; __syncthreads();
        for (int s = 128; s > 0; s >>= 1) { if (t < s) red[t] += red[t + s]; __syncthreads(); }
        float r1 = red[0]; __syncthreads();
        red[t] = s2; __syncthreads();
        for (int s = 128; s > 0; s >>= 1) { if (t < s) red[t] += red[t + s]; __syncthreads(); }
        if (t == 0) {
            atomicAdd(&sums[b * CH2 + ch], r1);
            atomicAdd(&sums[b * CH2 + halfc + ch], red[0]);
        }
    }
}

// ---------------------------------------------------------------------------
// Kernel 4: SE gate  (mean -> FC relu -> FC clamp)
// ---------------------------------------------------------------------------
__global__ __launch_bounds__(256) void k_se(const float* __restrict__ sums,
                                            const float* __restrict__ w1,
                                            const float* __restrict__ b1,
                                            const float* __restrict__ w2,
                                            const float* __restrict__ b2,
                                            float* __restrict__ gate) {
    __shared__ float mean[512];
    __shared__ float hid[128];
    const int b = blockIdx.x, t = threadIdx.x;
    for (int i = t; i < 512; i += 256) mean[i] = sums[b * 512 + i] * (1.0f / 4096.0f);
    __syncthreads();
    if (t < 128) {
        float s = b1[t];
        for (int i = 0; i < 512; ++i) s += w1[t * 512 + i] * mean[i];
        hid[t] = fmaxf(s, 0.0f);
    }
    __syncthreads();
    for (int ch = t; ch < 512; ch += 256) {
        float s = b2[ch];
        for (int j = 0; j < 128; ++j) s += w2[ch * 128 + j] * hid[j];
        gate[b * 512 + ch] = fminf(fmaxf(s, 0.0f), 1.0f);
    }
}

// ---------------------------------------------------------------------------
// helpers
// ---------------------------------------------------------------------------
__global__ __launch_bounds__(256) void k_cvt(const float* __restrict__ src,
                                             _Float16* __restrict__ dst, int n) {
    int i = blockIdx.x * 256 + threadIdx.x;
    if (i < n) dst[i] = (_Float16)src[i];
}

__global__ __launch_bounds__(256) void k_zero(float* __restrict__ p, int n) {
    int i = blockIdx.x * 256 + threadIdx.x;
    if (i < n) p[i] = 0.0f;
}

// ---------------------------------------------------------------------------
extern "C" void kernel_launch(void* const* d_in, const int* in_sizes, int n_in,
                              void* d_out, int out_size, void* d_ws, size_t ws_size,
                              hipStream_t stream) {
    const float* x     = (const float*)d_in[0];
    const float* dw_w  = (const float*)d_in[1];
    const float* dw_g  = (const float*)d_in[2];
    const float* dw_b  = (const float*)d_in[3];
    const float* g1p_w = (const float*)d_in[4];
    const float* g1p_g = (const float*)d_in[5];
    const float* g1p_b = (const float*)d_in[6];
    const float* g1c_w = (const float*)d_in[7];
    const float* g1c_g = (const float*)d_in[8];
    const float* g1c_b = (const float*)d_in[9];
    const float* g2p_w = (const float*)d_in[10];
    const float* g2p_g = (const float*)d_in[11];
    const float* g2p_b = (const float*)d_in[12];
    const float* g2c_w = (const float*)d_in[13];
    const float* g2c_g = (const float*)d_in[14];
    const float* g2c_b = (const float*)d_in[15];
    const float* se_w1 = (const float*)d_in[16];
    const float* se_b1 = (const float*)d_in[17];
    const float* se_w2 = (const float*)d_in[18];
    const float* se_b2 = (const float*)d_in[19];
    const float* g3p_w = (const float*)d_in[20];
    const float* g3p_g = (const float*)d_in[21];
    const float* g3p_b = (const float*)d_in[22];
    const float* g3c_w = (const float*)d_in[23];
    const float* g3c_g = (const float*)d_in[24];
    const float* g3c_b = (const float*)d_in[25];
    float* outp = (float*)d_out;

    // workspace layout
    char* w = (char*)d_ws;
    size_t o = 0;
    _Float16* a_h  = (_Float16*)(w + o); o += (size_t)16 * 128 * 4096 * 2;   // attn out f16
    float*    bufA = (float*)   (w + o); o += (size_t)16 * 256 * 4096 * 4;   // ghost-primary x1 (reused)
    float*    yv   = (float*)   (w + o); o += (size_t)16 * 128 * 4096 * 4;   // y = mosa + x
    _Float16* y_h  = (_Float16*)(w + o); o += (size_t)16 * 128 * 4096 * 2;
    _Float16* h_h  = (_Float16*)(w + o); o += (size_t)16 * 512 * 4096 * 2;   // ghost2 out f16
    _Float16* w1h  = (_Float16*)(w + o); o += (size_t)64 * 128 * 2;
    _Float16* w2h  = (_Float16*)(w + o); o += (size_t)256 * 128 * 2;
    _Float16* w3h  = (_Float16*)(w + o); o += (size_t)64 * 512 * 2;
    float*    sums = (float*)   (w + o); o += (size_t)16 * 512 * 4;
    float*    gate = (float*)   (w + o); o += (size_t)16 * 512 * 4;

    // weight conversions + SE accumulator zero
    k_cvt<<<(8192 + 255) / 256, 256, 0, stream>>>(g1p_w, w1h, 8192);
    k_cvt<<<(32768 + 255) / 256, 256, 0, stream>>>(g2p_w, w2h, 32768);
    k_cvt<<<(32768 + 255) / 256, 256, 0, stream>>>(g3p_w, w3h, 32768);
    k_zero<<<(8192 + 255) / 256, 256, 0, stream>>>(sums, 8192);

    // MoSA attention + dwconv
    k_attn<<<2048, 256, 0, stream>>>(x, dw_w, dw_g, dw_b, a_h);

    // Ghost1: 128 -> 64 primary, cheap, concat + residual x  -> y
    k_gemm<<<512, 256, (64 * 128 + 128 * 128) * 2, stream>>>(a_h, w1h, g1p_g, g1p_b, nullptr, bufA, 128, 64);
    k_cheap<<<16 * 64, 256, 0, stream>>>(bufA, g1c_w, g1c_g, g1c_b, 64, x, 1.0f, yv, y_h, nullptr);

    // Ghost2: 128 -> 256 primary, cheap -> h (f16) + SE sums
    k_gemm<<<512, 256, (256 * 128 + 128 * 128) * 2, stream>>>(y_h, w2h, g2p_g, g2p_b, nullptr, bufA, 128, 256);
    k_cheap<<<16 * 256, 256, 0, stream>>>(bufA, g2c_w, g2c_g, g2c_b, 256, nullptr, 0.0f, nullptr, h_h, sums);

    // SE gate
    k_se<<<16, 256, 0, stream>>>(sums, se_w1, se_b1, se_w2, se_b2, gate);

    // Ghost3: 512 -> 64 primary (gate folded into staging), cheap, out = f + 2y
    k_gemm<<<512, 256, (64 * 512 + 512 * 128) * 2, stream>>>(h_h, w3h, g3p_g, g3p_b, gate, bufA, 512, 64);
    k_cheap<<<16 * 64, 256, 0, stream>>>(bufA, g3c_w, g3c_g, g3c_b, 64, yv, 2.0f, outp, nullptr, nullptr);
}
